// DropoutRNN_87153476370512
// MI455X (gfx1250) — matrix operand order repaired and checked by reading, not compiled, not run on hardware
//
#include <hip/hip_runtime.h>
#include <math.h>
#include <stdint.h>

// ---------------------------------------------------------------------------
// DropoutRNN for MI455X (gfx1250), wave32 + WMMA bf16 + async global->LDS.
//   h_{s+1} = tanh( x_s @ Wi^T + bi + (0.5*h_s) @ Wh^T )
// One fused K=1536 GEMM per step; weights pre-swizzled to the bf16 WMMA
// B-fragment layout; per k-block the workgroup double-buffers B through LDS
// with GLOBAL_LOAD_ASYNC_TO_LDS_B128 (ASYNCcnt) so weight fetch overlaps
// the matrix pipe, and each LDS block is shared by 4 m-tiles (4x L2 cut).
// ---------------------------------------------------------------------------

typedef __attribute__((ext_vector_type(16))) __bf16 v16bf;
typedef __attribute__((ext_vector_type(8)))  float  v8f;

#define S_LEN 512
#define B_SZ  256
#define I_SZ  512
#define H_SZ  1024
#define K_TOT (I_SZ + H_SZ)     // 1536
#define K_BLKS (K_TOT / 32)     // 48 k-steps of 32
#define BLK_BYTES 1024          // one 32x16 bf16 B block = 32 lanes * 32 B
#define NBLK_PER_WG 16          // 256 columns per workgroup
#define STAGE_BYTES (NBLK_PER_WG * BLK_BYTES)   // 16 KB per k-step

// ---------------------------------------------------------------------------
// One-time weight pack: W[k][n] (k<512 -> Wi[n][k], else Wh[n][k-512]) into
// bf16 blocks [n_blk][k_blk][lane][16] in this file's contiguous-k order for the 16x16x32 bf16 B
// fragment (same k order as A): lanes 0-15 hold K=0-15 (elem e <-> K=e), lanes 16-31 hold
// K=16-31; N = lane&15. Each lane's 16 values are contiguous (32 B).
// ---------------------------------------------------------------------------
__global__ void wswz_kernel(const float* __restrict__ Wi,
                            const float* __restrict__ Wh,
                            __bf16* __restrict__ out) {
  int idx  = blockIdx.x * blockDim.x + threadIdx.x;   // < K_TOT*H_SZ
  int pos  = idx & 15;
  int lane = (idx >> 4) & 31;
  int blk  = idx >> 9;                                // nb*K_BLKS + kb
  int kb   = blk % K_BLKS;
  int nb   = blk / K_BLKS;
  int kloc = ((lane >> 4) << 4) + pos;                // kg*16 + e
  int k    = kb * 32 + kloc;
  int n    = nb * 16 + (lane & 15);                   // output unit h
  float v  = (k < I_SZ) ? Wi[(size_t)n * I_SZ + k]
                        : Wh[(size_t)n * H_SZ + (k - I_SZ)];
  out[idx] = (__bf16)v;
}

// async global -> LDS, 16 bytes per lane, ASYNCcnt-tracked
__device__ __forceinline__ void async_copy_b128(unsigned lds_off,
                                                const void* gaddr) {
  asm volatile("global_load_async_to_lds_b128 %0, %1, off"
               :: "v"(lds_off), "v"((unsigned long long)(uintptr_t)gaddr)
               : "memory");
}

__device__ __forceinline__ void wait_asynccnt0() {
  asm volatile("s_wait_asynccnt 0x0" ::: "memory");
}

// ---------------------------------------------------------------------------
// One recurrence step. Grid: (4, 4) blocks x 512 threads (16 waves).
// Block covers 4 m-tiles (64 rows) x 256 columns. Wave (mt,ng) -> 16x64 tile.
// B staged through double-buffered LDS; each staged block feeds 4 waves.
// ---------------------------------------------------------------------------
__global__ __launch_bounds__(512)
void rnn_step_kernel(const float* __restrict__ x,      // [S,B,I]
                     const float* __restrict__ hprev,  // [B,H] f32
                     float*       __restrict__ hnext,  // [B,H] f32
                     const __bf16* __restrict__ wswz,  // packed weights
                     const float* __restrict__ bi,     // [H]
                     int s) {
  __shared__ __align__(32) unsigned char sbuf[2][STAGE_BYTES];

  const int tid     = threadIdx.x;
  const int lane    = tid & 31;
  const int wave    = tid >> 5;          // 0..15
  const int mt_loc  = wave & 3;          // 0..3
  const int ng_loc  = wave >> 2;         // 0..3 (64-col groups in block)
  const int m_tile  = blockIdx.x * 4 + mt_loc;      // 0..15
  const int nb0     = blockIdx.y * NBLK_PER_WG;     // first 16-col block

  const int m   = lane & 15;
  const int kg  = lane >> 4;             // lane-half selects K phase
  const int row = m_tile * 16 + m;       // batch index

  const float* xrow = x + ((size_t)s * B_SZ + row) * I_SZ;
  const float* hrow = hprev + (size_t)row * H_SZ;
  const unsigned char* wbytes = (const unsigned char*)wswz;

  // cooperative async stage of one k-block (16 KB) into sbuf[b]
  const int c0  = tid * 2;               // two 16B chunks per thread
  auto stage = [&](int kb, int b) {
#pragma unroll
    for (int i = 0; i < 2; ++i) {
      int c   = c0 + i;                  // 0..1023
      int nbl = c >> 6;                  // 0..15: which 16-col block
      int co  = (c & 63) * 16;           // byte offset inside block
      const unsigned char* g =
          wbytes + ((size_t)(nb0 + nbl) * K_BLKS + kb) * BLK_BYTES + co;
      unsigned loff = (unsigned)(uintptr_t)&sbuf[b][nbl * BLK_BYTES + co];
      async_copy_b128(loff, g);
    }
  };

  v8f acc[4] = {};

  stage(0, 0);
  wait_asynccnt0();
  __syncthreads();

  for (int kb = 0; kb < K_BLKS; ++kb) {
    const int cur = kb & 1;
    if (kb + 1 < K_BLKS) stage(kb + 1, cur ^ 1);   // overlap with compute

    // ---- build A fragment (16x32 bf16, contiguous k = kg*16+e) with on-the-fly cvt ----
    const float* src;
    float scale;
    if (kb < I_SZ / 32) { src = xrow + kb * 32;               scale = 1.0f; }
    else                { src = hrow + (kb - I_SZ / 32) * 32; scale = 0.5f; }
    const float4* p = (const float4*)(src + kg * 16);
    float4 f0 = p[0], f1 = p[1], f2 = p[2], f3 = p[3];
    float u[16] = { f0.x, f0.y, f0.z, f0.w, f1.x, f1.y, f1.z, f1.w,
                    f2.x, f2.y, f2.z, f2.w, f3.x, f3.y, f3.z, f3.w };
    v16bf a;
#pragma unroll
    for (int e = 0; e < 16; ++e) a[e] = (__bf16)(u[e] * scale);

    // ---- B fragments from LDS (ds_load_b128 x2 per tile) + 4 WMMAs ----
#pragma unroll
    for (int t = 0; t < 4; ++t) {
      const v16bf b =
          *(const v16bf*)&sbuf[cur][(ng_loc * 4 + t) * BLK_BYTES + lane * 32];
      acc[t] = __builtin_amdgcn_wmma_f32_16x16x32_bf16(
          false, a, false, b, (short)0, acc[t], false, false);
    }

    wait_asynccnt0();     // next buffer resident (this wave's copies done)
    __syncthreads();      // all waves done reading cur / writing nxt
  }

  // ---- epilogue: bias + tanh, C/D layout (VGPR r -> row kg*8+r) ----
  const int ncol = lane & 15;
#pragma unroll
  for (int t = 0; t < 4; ++t) {
    const int n_out = nb0 * 16 + ng_loc * 64 + t * 16 + ncol;
    const float bv = bi[n_out];
#pragma unroll
    for (int r = 0; r < 8; ++r) {
      const int m_out = m_tile * 16 + kg * 8 + r;
      hnext[(size_t)m_out * H_SZ + n_out] = tanhf(acc[t][r] + bv);
    }
  }
}

// ---------------------------------------------------------------------------
extern "C" void kernel_launch(void* const* d_in, const int* in_sizes, int n_in,
                              void* d_out, int out_size, void* d_ws, size_t ws_size,
                              hipStream_t stream) {
  const float* x  = (const float*)d_in[0];
  const float* h0 = (const float*)d_in[1];
  const float* Wi = (const float*)d_in[2];
  const float* bi = (const float*)d_in[3];
  const float* Wh = (const float*)d_in[4];

  char*   ws   = (char*)d_ws;
  __bf16* wswz = (__bf16*)ws;                                   // 3 MiB
  float*  h_a  = (float*)(ws + (size_t)K_TOT * H_SZ * sizeof(__bf16));
  float*  h_b  = h_a + (size_t)B_SZ * H_SZ;                     // 1 MiB each

  // h state init from h0 (device-to-device, capture-safe)
  hipMemcpyAsync(h_a, h0, (size_t)B_SZ * H_SZ * sizeof(float),
                 hipMemcpyDeviceToDevice, stream);

  // pack weights to bf16 WMMA-B layout once
  wswz_kernel<<<(K_TOT * H_SZ) / 256, 256, 0, stream>>>(Wi, Wh, wswz);

  // sequential recurrence; last step writes d_out directly
  float* prev = h_a;
  float* next = h_b;
  for (int s = 0; s < S_LEN; ++s) {
    float* dst = (s == S_LEN - 1) ? (float*)d_out : next;
    rnn_step_kernel<<<dim3(4, 4), 512, 0, stream>>>(x, prev, dst, wswz, bi, s);
    next = prev;
    prev = dst;
  }
}
